// DGCNN_FAPE_33663953666361
// MI455X (gfx1250) — matrix operand order repaired and checked by reading, not compiled, not run on hardware
//
#include <hip/hip_runtime.h>
#include <hip/hip_fp16.h>
#include <hip/hip_bf16.h>

// ---------------- CDNA5 WMMA vector types (probe-verified) ----------------
typedef __attribute__((ext_vector_type(16))) _Float16 v16h;
typedef __attribute__((ext_vector_type(8)))  float    v8f;

// ---------------- problem constants (from reference) ----------------------
constexpr int Bc   = 4;
constexpr int Lc   = 7;
constexpr int Kc   = 16;
constexpr int Gc   = 4;
constexpr float EPSc = 1e-5f;
constexpr float NEGc = 0.2f;
constexpr int Nfull = 14336;   // points per batch
constexpr int N1    = 3584;    // after first FPS
constexpr int N2    = 896;     // after second FPS
constexpr int nA    = Nfull / Lc; // 2048 per group
constexpr int nB    = N1 / Lc;    // 512 per group
constexpr int nC    = N2 / Lc;    // 128 per group

// ======================================================================
// Kernel 0: f0 = W_in @ x + b_in   (10x10 per-point MLP)
// ======================================================================
__global__ void input_mlp_kernel(const float* __restrict__ x,
                                 const float* __restrict__ W,
                                 const float* __restrict__ bias,
                                 float* __restrict__ f0) {
    int t = blockIdx.x * blockDim.x + threadIdx.x;
    if (t >= Bc * Nfull) return;
    int n = t % Nfull, b = t / Nfull;
    float xi[10];
    #pragma unroll
    for (int c = 0; c < 10; ++c) xi[c] = x[((size_t)b * 10 + c) * Nfull + n];
    #pragma unroll
    for (int o = 0; o < 10; ++o) {
        float s = bias[o];
        #pragma unroll
        for (int c = 0; c < 10; ++c) s += W[o * 10 + c] * xi[c];
        f0[((size_t)b * 10 + o) * Nfull + n] = s;
    }
}

// ======================================================================
// Group-local kNN (K=16). One thread per query; keys staged in LDS.
// idx_out[b][gq][k] = group-local key index (ascending distance).
// ======================================================================
__global__ void knn_kernel(const float* __restrict__ coorQ, int Mq,
                           const float* __restrict__ coorK, int Mk,
                           int nq, int nk, int* __restrict__ idx_out) {
    __shared__ float kx[2048], ky[2048], kz[2048];
    const int blocksPerGroup = (nq + 255) / 256;
    int tile = blockIdx.x % blocksPerGroup;
    int bl   = blockIdx.x / blocksPerGroup;
    int l = bl % Lc, b = bl / Lc;

    for (int i = threadIdx.x; i < nk; i += blockDim.x) {
        int gk = l * nk + i;
        kx[i] = coorK[((size_t)b * 10 + 0) * Mk + gk];
        ky[i] = coorK[((size_t)b * 10 + 1) * Mk + gk];
        kz[i] = coorK[((size_t)b * 10 + 2) * Mk + gk];
    }
    __syncthreads();

    int q = tile * 256 + threadIdx.x;
    if (q >= nq) return;
    int gq = l * nq + q;
    float qx = coorQ[((size_t)b * 10 + 0) * Mq + gq];
    float qy = coorQ[((size_t)b * 10 + 1) * Mq + gq];
    float qz = coorQ[((size_t)b * 10 + 2) * Mq + gq];

    float bd[Kc]; int bi[Kc];
    #pragma unroll
    for (int k = 0; k < Kc; ++k) { bd[k] = 1e30f; bi[k] = 0; }

    for (int i = 0; i < nk; ++i) {
        float dx = qx - kx[i], dy = qy - ky[i], dz = qz - kz[i];
        float d = dx * dx + dy * dy + dz * dz;
        if (d < bd[Kc - 1]) {               // insertion keeps ascending order,
            int p = Kc - 1;                 // stable on ties (first occurrence)
            while (p > 0 && bd[p - 1] > d) { bd[p] = bd[p - 1]; bi[p] = bi[p - 1]; --p; }
            bd[p] = d; bi[p] = i;
        }
    }
    int* o = idx_out + ((size_t)b * Mq + gq) * Kc;
    #pragma unroll
    for (int k = 0; k < Kc; ++k) o[k] = bi[k];
}

// ======================================================================
// Fused edge-feature + GEMM tile via v_wmma_f32_16x16x32_f16.
// One wave handles one (query q, 16-wide cout tile ct):
//   M (16 rows)  = the query's 16 kNN edges
//   N (16 cols)  = cout channels [ct*16, ct*16+16)
//   K            = CIN (edge channels = [fk-fq, fq]), padded to mult of 32
// Operand packing per CDNA5 ISA 7.12.2 (wave32):
//   A 16x32 f16: lane m=l&15, hi=l>>4; VGPR v holds K = (v&3)*2+(v>>2)*16+hi*8
//   B 32x16 f16: lane ncol=l&15; VGPR j holds K = (j&3)*2+(j>>2)*16+hi*8
// ======================================================================
template <int CIN, int COUT>
__device__ inline void edge_wmma_acc(const float* __restrict__ fq,
                                     const float* __restrict__ fk,
                                     const float* __restrict__ W,
                                     const int* __restrict__ idx,
                                     int b, int q, int ct,
                                     int Nq, int Mk, int nq, int nk,
                                     v8f& acc) {
    constexpr int C = CIN / 2;
    constexpr int KSTEPS = (CIN + 31) / 32;
    const int lane = threadIdx.x & 31;
    const int m  = lane & 15;     // edge (neighbor) index within query
    const int hi = lane >> 4;
    const int l  = q / nq;        // label group
    const int j  = idx[((size_t)b * Nq + q) * Kc + m];  // group-local key
    const int jg = l * nk + j;                          // global key position

    __builtin_prefetch(W, 0, 3);  // -> global_prefetch_b8 on gfx1250

    auto edgeval = [&](int c) -> float {
        if (c >= CIN) return 0.f;                       // zero pad K dim
        if (c < C)
            return fk[((size_t)b * C + c) * Mk + jg] -
                   fq[((size_t)b * C + c) * Nq + q];    // gathered - center
        return fq[((size_t)b * C + (c - C)) * Nq + q];  // center
    };

    #pragma unroll
    for (int ks = 0; ks < KSTEPS; ++ks) {
        v16h a;
        #pragma unroll
        for (int v = 0; v < 8; ++v) {
            int c0 = ks * 32 + (v & 3) * 2 + (v >> 2) * 16 + hi * 8;
            a[2 * v]     = (_Float16)edgeval(c0);
            a[2 * v + 1] = (_Float16)edgeval(c0 + 1);
        }
        v16h bm;
        const int orow = ct * 16 + m;   // output channel (row of W)
        #pragma unroll
        for (int jv = 0; jv < 8; ++jv) {
            int k0 = ks * 32 + (jv & 3) * 2 + (jv >> 2) * 16 + hi * 8;
            bm[2 * jv]     = (_Float16)((k0     < CIN) ? W[orow * CIN + k0]     : 0.f);
            bm[2 * jv + 1] = (_Float16)((k0 + 1 < CIN) ? W[orow * CIN + k0 + 1] : 0.f);
        }
        acc = __builtin_amdgcn_wmma_f32_16x16x32_f16(
            /*neg_a=*/false, a, /*neg_b=*/false, bm,
            /*c_mod=*/(short)0, acc, /*reuse_a=*/false, /*reuse_b=*/false);
    }
}

// Pass 1: accumulate group-norm sums (sum, sumsq) per (batch, channel-group).
template <int CIN, int COUT>
__global__ void edgeconv_stats_kernel(const float* __restrict__ fq,
                                      const float* __restrict__ fk,
                                      const float* __restrict__ W,
                                      const int* __restrict__ idx,
                                      float* __restrict__ stats,
                                      int Nq, int Mk, int nq, int nk) {
    constexpr int TILES = COUT / 16;
    const int w  = blockIdx.x;           // one wave per block (blockDim=32)
    const int ct = w % TILES;
    const int bq = w / TILES;
    const int b  = bq / Nq;
    const int q  = bq % Nq;

    v8f acc = {};
    edge_wmma_acc<CIN, COUT>(fq, fk, W, idx, b, q, ct, Nq, Mk, nq, nk, acc);

    const int lane = threadIdx.x;
    const int cout = ct * 16 + (lane & 15);
    const int g    = cout / (COUT / Gc);
    float s = 0.f, s2 = 0.f;
    #pragma unroll
    for (int r = 0; r < 8; ++r) { float v = acc[r]; s += v; s2 += v * v; }

    __shared__ float ssum[Gc], ssq[Gc];
    if (lane < Gc) { ssum[lane] = 0.f; ssq[lane] = 0.f; }
    __syncthreads();
    atomicAdd(&ssum[g], s);
    atomicAdd(&ssq[g], s2);
    __syncthreads();
    if (lane < Gc) {
        atomicAdd(&stats[(b * Gc + lane) * 2 + 0], ssum[lane]);
        atomicAdd(&stats[(b * Gc + lane) * 2 + 1], ssq[lane]);
    }
}

// Pass 2: recompute tile, normalize, LeakyReLU, max over the 16 neighbors
// (M-dimension lives in the 8 D-VGPRs + lane^16), write f[b][cout][q].
template <int CIN, int COUT>
__global__ void edgeconv_out_kernel(const float* __restrict__ fq,
                                    const float* __restrict__ fk,
                                    const float* __restrict__ W,
                                    const float* __restrict__ gamma,
                                    const float* __restrict__ beta,
                                    const int* __restrict__ idx,
                                    const float* __restrict__ stats,
                                    float* __restrict__ out,
                                    int Nq, int Mk, int nq, int nk) {
    constexpr int TILES = COUT / 16;
    const int w  = blockIdx.x;
    const int ct = w % TILES;
    const int bq = w / TILES;
    const int b  = bq / Nq;
    const int q  = bq % Nq;

    v8f acc = {};
    edge_wmma_acc<CIN, COUT>(fq, fk, W, idx, b, q, ct, Nq, Mk, nq, nk, acc);

    const int lane = threadIdx.x;
    const int cout = ct * 16 + (lane & 15);
    const int g    = cout / (COUT / Gc);
    const float cnt = (float)Nq * (float)Kc * (float)(COUT / Gc);
    float su = stats[(b * Gc + g) * 2 + 0];
    float sq = stats[(b * Gc + g) * 2 + 1];
    float mu  = su / cnt;
    float var = sq / cnt - mu * mu;
    float rs  = rsqrtf(var + EPSc);
    float ga = gamma[cout], be = beta[cout];

    float mx = -1e30f;
    #pragma unroll
    for (int r = 0; r < 8; ++r) {
        float hn = (acc[r] - mu) * rs * ga + be;
        hn = (hn >= 0.f) ? hn : NEGc * hn;
        mx = fmaxf(mx, hn);
    }
    mx = fmaxf(mx, __shfl_xor(mx, 16, 32));   // combine the two M-halves
    if ((lane >> 4) == 0)
        out[((size_t)b * COUT + cout) * Nq + q] = mx;
}

// ======================================================================
// Farthest point sampling: one block per (b,l), sequential like reference.
// ======================================================================
__global__ void fps_kernel(const float* __restrict__ coor, int M,
                           int n, int m, int num, int* __restrict__ glob) {
    __shared__ float px[2048], py[2048], pz[2048], dist[2048];
    __shared__ float rmax[256];
    __shared__ int   ridx[256];
    const int l = blockIdx.x % Lc, b = blockIdx.x / Lc;
    const int tid = threadIdx.x;

    for (int i = tid; i < n; i += 256) {
        int gi = l * n + i;
        px[i] = coor[((size_t)b * 10 + 0) * M + gi];
        py[i] = coor[((size_t)b * 10 + 1) * M + gi];
        pz[i] = coor[((size_t)b * 10 + 2) * M + gi];
    }
    __syncthreads();
    for (int i = tid; i < n; i += 256) {
        float dx = px[i] - px[0], dy = py[i] - py[0], dz = pz[i] - pz[0];
        dist[i] = dx * dx + dy * dy + dz * dz;
    }
    if (tid == 0) glob[(size_t)b * num + l * m] = l * n;
    __syncthreads();

    for (int t = 1; t < m; ++t) {
        // argmax(dist), first occurrence on ties
        float best = -1e30f; int bidx = 0x7fffffff;
        for (int i = tid; i < n; i += 256)
            if (dist[i] > best) { best = dist[i]; bidx = i; }
        rmax[tid] = best; ridx[tid] = bidx;
        __syncthreads();
        for (int s = 128; s > 0; s >>= 1) {
            if (tid < s) {
                if (rmax[tid + s] > rmax[tid] ||
                    (rmax[tid + s] == rmax[tid] && ridx[tid + s] < ridx[tid])) {
                    rmax[tid] = rmax[tid + s]; ridx[tid] = ridx[tid + s];
                }
            }
            __syncthreads();
        }
        int nxt = ridx[0];
        if (tid == 0) glob[(size_t)b * num + l * m + t] = l * n + nxt;
        float cx = px[nxt], cy = py[nxt], cz = pz[nxt];
        __syncthreads();
        for (int i = tid; i < n; i += 256) {
            float dx = px[i] - cx, dy = py[i] - cy, dz = pz[i] - cz;
            dist[i] = fminf(dist[i], dx * dx + dy * dy + dz * dz);
        }
        __syncthreads();
    }
}

// gather out[b][c][i] = in[b][c][glob[b][i]]
__global__ void gather_kernel(const float* __restrict__ in,
                              const int* __restrict__ glob,
                              float* __restrict__ out,
                              int Cch, int Min, int num) {
    size_t t = (size_t)blockIdx.x * blockDim.x + threadIdx.x;
    size_t total = (size_t)Bc * Cch * num;
    if (t >= total) return;
    int i = (int)(t % num);
    int c = (int)((t / num) % Cch);
    int b = (int)(t / ((size_t)num * Cch));
    out[((size_t)b * Cch + c) * num + i] =
        in[((size_t)b * Cch + c) * Min + glob[(size_t)b * num + i]];
}

__global__ void copy_i32_kernel(const int* __restrict__ in,
                                int* __restrict__ outp, int n) {
    int t = blockIdx.x * blockDim.x + threadIdx.x;
    if (t < n) outp[t] = in[t];
}

// ======================================================================
// Host orchestration
// ======================================================================
extern "C" void kernel_launch(void* const* d_in, const int* in_sizes, int n_in,
                              void* d_out, int out_size, void* d_ws, size_t ws_size,
                              hipStream_t stream) {
    (void)in_sizes; (void)n_in; (void)out_size; (void)ws_size;
    const float* x    = (const float*)d_in[0];
    const float* W_in = (const float*)d_in[1];
    const float* b_in = (const float*)d_in[2];
    const float* W1   = (const float*)d_in[3];
    const float* g1   = (const float*)d_in[4];
    const float* be1  = (const float*)d_in[5];
    const float* W2   = (const float*)d_in[6];
    const float* g2   = (const float*)d_in[7];
    const float* be2  = (const float*)d_in[8];
    const float* W3   = (const float*)d_in[9];
    const float* g3   = (const float*)d_in[10];
    const float* be3  = (const float*)d_in[11];
    const float* W4   = (const float*)d_in[12];
    const float* g4   = (const float*)d_in[13];
    const float* be4  = (const float*)d_in[14];
    float* out = (float*)d_out;

    // ---- workspace carve-out (all 4-byte elements) ----
    float* ws = (float*)d_ws;
    size_t off = 0;
    auto alloc = [&](size_t nelem) { float* p = ws + off; off += nelem; return p; };
    float* f0    = alloc((size_t)Bc * 10 * Nfull);
    float* f1    = alloc((size_t)Bc * 32 * Nfull);
    float* f2    = alloc((size_t)Bc * 64 * N1);
    float* f3    = alloc((size_t)Bc * 64 * N1);
    float* coor1 = alloc((size_t)Bc * 10 * N1);
    float* fq1   = alloc((size_t)Bc * 32 * N1);
    float* fq2   = alloc((size_t)Bc * 64 * N2);
    float* stats = alloc((size_t)Bc * Gc * 2);
    int* glob1 = (int*)alloc((size_t)Bc * N1);
    int* glob2 = (int*)alloc((size_t)Bc * N2);
    int* idxA  = (int*)alloc((size_t)Bc * Nfull * Kc);
    int* idxB  = (int*)alloc((size_t)Bc * N1 * Kc);
    int* idxC  = (int*)alloc((size_t)Bc * N1 * Kc);
    int* idxD  = (int*)alloc((size_t)Bc * N2 * Kc);

    float* out_coor = out;                                   // (B,10,896)
    float* out_f    = out + (size_t)Bc * 10 * N2;            // (B,128,896)
    int*   out_idx1 = (int*)(out + (size_t)Bc * 10 * N2 + (size_t)Bc * 128 * N2);
    int*   out_idx2 = out_idx1 + Bc * N1;

    const size_t statsBytes = (size_t)Bc * Gc * 2 * sizeof(float);

    // ---- stage 0: input MLP ----
    input_mlp_kernel<<<(Bc * Nfull + 255) / 256, 256, 0, stream>>>(x, W_in, b_in, f0);

    // ---- stage 1: knn(x,x) + edgeconv<20,32> -> f1 ----
    knn_kernel<<<Bc * Lc * ((nA + 255) / 256), 256, 0, stream>>>(x, Nfull, x, Nfull, nA, nA, idxA);
    hipMemsetAsync(stats, 0, statsBytes, stream);
    {
        int grid = Bc * Nfull * (32 / 16);
        edgeconv_stats_kernel<20, 32><<<grid, 32, 0, stream>>>(
            f0, f0, W1, idxA, stats, Nfull, Nfull, nA, nA);
        edgeconv_out_kernel<20, 32><<<grid, 32, 0, stream>>>(
            f0, f0, W1, g1, be1, idxA, stats, f1, Nfull, Nfull, nA, nA);
    }

    // ---- FPS 1: 2048 -> 512 per group; gather coords + feats ----
    fps_kernel<<<Bc * Lc, 256, 0, stream>>>(x, Nfull, nA, nB, N1, glob1);
    gather_kernel<<<(int)(((size_t)Bc * 10 * N1 + 255) / 256), 256, 0, stream>>>(
        x, glob1, coor1, 10, Nfull, N1);
    gather_kernel<<<(int)(((size_t)Bc * 32 * N1 + 255) / 256), 256, 0, stream>>>(
        f1, glob1, fq1, 32, Nfull, N1);

    // ---- stage 2: knn(coor1,x) + edgeconv<64,64> -> f2 ----
    knn_kernel<<<Bc * Lc * ((nB + 255) / 256), 256, 0, stream>>>(coor1, N1, x, Nfull, nB, nA, idxB);
    hipMemsetAsync(stats, 0, statsBytes, stream);
    {
        int grid = Bc * N1 * (64 / 16);
        edgeconv_stats_kernel<64, 64><<<grid, 32, 0, stream>>>(
            fq1, f1, W2, idxB, stats, N1, Nfull, nB, nA);
        edgeconv_out_kernel<64, 64><<<grid, 32, 0, stream>>>(
            fq1, f1, W2, g2, be2, idxB, stats, f2, N1, Nfull, nB, nA);
    }

    // ---- stage 3: knn(coor1,coor1) + edgeconv<128,64> -> f3 ----
    knn_kernel<<<Bc * Lc * ((nB + 255) / 256), 256, 0, stream>>>(coor1, N1, coor1, N1, nB, nB, idxC);
    hipMemsetAsync(stats, 0, statsBytes, stream);
    {
        int grid = Bc * N1 * (64 / 16);
        edgeconv_stats_kernel<128, 64><<<grid, 32, 0, stream>>>(
            f2, f2, W3, idxC, stats, N1, N1, nB, nB);
        edgeconv_out_kernel<128, 64><<<grid, 32, 0, stream>>>(
            f2, f2, W3, g3, be3, idxC, stats, f3, N1, N1, nB, nB);
    }

    // ---- FPS 2: 512 -> 128 per group; gather coords (straight to d_out) + feats ----
    fps_kernel<<<Bc * Lc, 256, 0, stream>>>(coor1, N1, nB, nC, N2, glob2);
    gather_kernel<<<(int)(((size_t)Bc * 10 * N2 + 255) / 256), 256, 0, stream>>>(
        coor1, glob2, out_coor, 10, N1, N2);
    gather_kernel<<<(int)(((size_t)Bc * 64 * N2 + 255) / 256), 256, 0, stream>>>(
        f3, glob2, fq2, 64, N1, N2);

    // ---- stage 4: knn(coor2,coor1) + edgeconv<128,128> -> d_out f region ----
    knn_kernel<<<Bc * Lc * ((nC + 255) / 256), 256, 0, stream>>>(out_coor, N2, coor1, N1, nC, nB, idxD);
    hipMemsetAsync(stats, 0, statsBytes, stream);
    {
        int grid = Bc * N2 * (128 / 16);
        edgeconv_stats_kernel<128, 128><<<grid, 32, 0, stream>>>(
            fq2, f3, W4, idxD, stats, N2, N1, nC, nB);
        edgeconv_out_kernel<128, 128><<<grid, 32, 0, stream>>>(
            fq2, f3, W4, g4, be4, idxD, stats, out_f, N2, N1, nC, nB);
    }

    // ---- index outputs ----
    copy_i32_kernel<<<(Bc * N1 + 255) / 256, 256, 0, stream>>>(glob1, out_idx1, Bc * N1);
    copy_i32_kernel<<<(Bc * N2 + 255) / 256, 256, 0, stream>>>(glob2, out_idx2, Bc * N2);
}
